// CausalSelfAttention_33243046871409
// MI455X (gfx1250) — matrix operand order repaired and checked by reading, not compiled, not run on hardware
//
#include <hip/hip_runtime.h>
#include <hip/hip_bf16.h>
#include <math.h>

typedef _Float16 v16h __attribute__((ext_vector_type(16)));
typedef _Float16 v8h  __attribute__((ext_vector_type(8)));
typedef float    v8f  __attribute__((ext_vector_type(8)));
typedef _Float16 h2   __attribute__((ext_vector_type(2)));
typedef int      v4i  __attribute__((ext_vector_type(4)));

#define EMBED 1024
#define HEADS 16
#define HDIM  64
#define SEQ   2048
#define BATCH 2
#define LDQKV 3072   // 3*EMBED

// gfx1250 async Global->LDS copy path (ASYNCcnt). Guarded: falls back to
// VGPR staging if this toolchain doesn't declare the builtins.
#if defined(__AMDGCN__) && __has_builtin(__builtin_amdgcn_global_load_async_to_lds_b128) && \
    __has_builtin(__builtin_amdgcn_s_wait_asynccnt)
#define USE_ASYNC_COPY 1
#else
#define USE_ASYNC_COPY 0
#endif

#if USE_ASYNC_COPY
typedef __attribute__((address_space(1))) v4i* as1_v4i_ptr;
typedef __attribute__((address_space(3))) v4i* as3_v4i_ptr;
__device__ __forceinline__ void async_copy16(const _Float16* gp, _Float16* lp) {
  __builtin_amdgcn_global_load_async_to_lds_b128(
      (as1_v4i_ptr)(_Float16*)gp, (as3_v4i_ptr)lp, 0, 0);
}
#endif

__device__ __forceinline__ v8f wmma_f16(v16h a, v16h b, v8f c) {
  // 8 args: (neg_a, A, neg_b, B, c_mod, C, reuse_a, reuse_b)
  return __builtin_amdgcn_wmma_f32_16x16x32_f16(false, a, false, b, (short)0, c,
                                                false, false);
}

// ---------------------------------------------------------------- casts
// 8 elements/thread: 2x b128 in, 1x b128 out.  All sizes divisible by 8.
__global__ void cast_f32_f16(const float* __restrict__ in,
                             _Float16* __restrict__ out, int n) {
  int i = (blockIdx.x * blockDim.x + threadIdx.x) * 8;
  if (i < n) {
    float4 a = *(const float4*)(in + i);
    float4 b = *(const float4*)(in + i + 4);
    v8h o;
    o[0] = (_Float16)a.x; o[1] = (_Float16)a.y; o[2] = (_Float16)a.z; o[3] = (_Float16)a.w;
    o[4] = (_Float16)b.x; o[5] = (_Float16)b.y; o[6] = (_Float16)b.z; o[7] = (_Float16)b.w;
    *(v8h*)(out + i) = o;
  }
}

// ---------------------------------------------------------------- GEMM
// C[M,N] = A[M,K] @ B[K,N] + bias[N].  128x64 tile per 256-thread block.
// 8 waves in a 4x2 grid; each wave owns a 32x32 register tile:
// 2 A-frags x 2 B-frags -> 4 WMMAs per 32-wide K step.
// Async path: lsA ping-pong, async DMA of A(k+1) and register-prefetched
// B(k+1) issued before the WMMAs of step k.
__global__ void wmma_gemm(const _Float16* __restrict__ A,
                          const _Float16* __restrict__ B,
                          const float* __restrict__ bias,
                          void* __restrict__ Cout,
                          int M, int N, int K, int halfOut) {
#if USE_ASYNC_COPY
  __shared__ __attribute__((aligned(16))) _Float16 lsA[2][128 * 32];  // [m][k] x2
#else
  __shared__ __attribute__((aligned(16))) _Float16 lsA[1][128 * 32];
#endif
  __shared__ __attribute__((aligned(16))) _Float16 lsB[64 * 32];      // [n][k] (transposed)
  const int tid = threadIdx.x;
  const int wave = tid >> 5, lane = tid & 31, g = lane >> 4, ln = lane & 15;
  const int mw = (wave & 3) * 2;   // first of 2 m-tiles (of 8)
  const int nw = (wave >> 2) * 2;  // first of 2 n-tiles (of 4)
  const int m0 = blockIdx.y * 128, n0 = blockIdx.x * 64;
  const int kk = tid >> 3, nblk = (tid & 7) * 8;  // B staging: row kk, 8 cols

  v8f acc[2][2] = {};

#if USE_ASYNC_COPY
  const int nk = K >> 5;
  {  // prologue: stage k-step 0
#pragma unroll
    for (int j = 0; j < 2; ++j) {
      int c = j * 256 + tid;                                   // 512 x 16B
      async_copy16(A + (size_t)(m0 + (c >> 2)) * K + (c & 3) * 8, &lsA[0][c * 8]);
    }
    v8h bv = *(const v8h*)&B[(size_t)kk * N + n0 + nblk];      // one b128 load
#pragma unroll
    for (int j = 0; j < 8; ++j) lsB[(nblk + j) * 32 + kk] = bv[j];
    __builtin_amdgcn_s_wait_asynccnt(0);
  }
  __syncthreads();

  for (int kt = 0; kt < nk; ++kt) {
    const int cur = kt & 1;
    const bool hasNext = (kt + 1) < nk;
    const int k0n = (kt + 1) << 5;
    v8h bv = {};
    if (hasNext) {  // overlap next-tile movement with this step's WMMAs
#pragma unroll
      for (int j = 0; j < 2; ++j) {
        int c = j * 256 + tid;
        async_copy16(A + (size_t)(m0 + (c >> 2)) * K + k0n + (c & 3) * 8,
                     &lsA[cur ^ 1][c * 8]);
      }
      bv = *(const v8h*)&B[(size_t)(k0n + kk) * N + n0 + nblk];
    }

    v16h af[2], bf[2];
#pragma unroll
    for (int d = 0; d < 8; ++d) {
      int blk = d >> 2, i = d & 3;
      int ka = blk * 16 + g * 8 + 2 * i;                       // A-frag K index
      int kb = blk * 16 + g * 8 + 2 * i;                       // B-frag K index
#pragma unroll
      for (int t = 0; t < 2; ++t) {
        h2 pa = *(const h2*)&lsA[cur][((mw + t) * 16 + ln) * 32 + ka];
        af[t][2 * d] = pa.x; af[t][2 * d + 1] = pa.y;
        h2 pb = *(const h2*)&lsB[((nw + t) * 16 + ln) * 32 + kb];
        bf[t][2 * d] = pb.x; bf[t][2 * d + 1] = pb.y;
      }
    }
#pragma unroll
    for (int i2 = 0; i2 < 2; ++i2)
#pragma unroll
      for (int j2 = 0; j2 < 2; ++j2)
        acc[i2][j2] = wmma_f16(af[i2], bf[j2], acc[i2][j2]);

    __syncthreads();                 // all waves done reading lsB / lsA[cur]
    if (hasNext) {
#pragma unroll
      for (int j = 0; j < 8; ++j) lsB[(nblk + j) * 32 + kk] = bv[j];
      __builtin_amdgcn_s_wait_asynccnt(0);
    }
    __syncthreads();                 // next tiles published
  }
#else
  for (int k0 = 0; k0 < K; k0 += 32) {
#pragma unroll
    for (int j = 0; j < 16; ++j) {
      int e = j * 256 + tid;
      lsA[0][e] = A[(size_t)(m0 + (e >> 5)) * K + k0 + (e & 31)];
    }
    v8h bv = *(const v8h*)&B[(size_t)(k0 + kk) * N + n0 + nblk];
#pragma unroll
    for (int j = 0; j < 8; ++j) lsB[(nblk + j) * 32 + kk] = bv[j];
    __syncthreads();

    v16h af[2], bf[2];
#pragma unroll
    for (int d = 0; d < 8; ++d) {
      int blk = d >> 2, i = d & 3;
      int ka = blk * 16 + g * 8 + 2 * i;
      int kb = blk * 16 + g * 8 + 2 * i;
#pragma unroll
      for (int t = 0; t < 2; ++t) {
        h2 pa = *(const h2*)&lsA[0][((mw + t) * 16 + ln) * 32 + ka];
        af[t][2 * d] = pa.x; af[t][2 * d + 1] = pa.y;
        h2 pb = *(const h2*)&lsB[((nw + t) * 16 + ln) * 32 + kb];
        bf[t][2 * d] = pb.x; bf[t][2 * d + 1] = pb.y;
      }
    }
#pragma unroll
    for (int i2 = 0; i2 < 2; ++i2)
#pragma unroll
      for (int j2 = 0; j2 < 2; ++j2)
        acc[i2][j2] = wmma_f16(af[i2], bf[j2], acc[i2][j2]);
    __syncthreads();
  }
#endif

#pragma unroll
  for (int i = 0; i < 2; ++i) {
#pragma unroll
    for (int jj = 0; jj < 2; ++jj) {
      int col = n0 + (nw + jj) * 16 + ln;
      float bv = bias[col];
#pragma unroll
      for (int v = 0; v < 8; ++v) {
        int row = m0 + (mw + i) * 16 + v + 8 * g;              // C layout: M = v + 8*g
        float o = acc[i][jj][v] + bv;
        if (halfOut) ((_Float16*)Cout)[(size_t)row * N + col] = (_Float16)o;
        else         ((float*)Cout)[(size_t)row * N + col]    = o;
      }
    }
  }
}

// ---------------------------------------------------------------- flash attention
// qkv: [B*T, 3C] f16 (Q | K | V head slices). ctx: [B*T, C] f16.
// Grid (T/64, H, B), 128 threads = 4 waves x 16 queries, key chunks of 32.
__global__ void attn_kernel(const _Float16* __restrict__ qkv,
                            _Float16* __restrict__ ctx) {
  __shared__ __attribute__((aligned(16))) _Float16 lsK[32 * 64];      // [key][d]
  __shared__ __attribute__((aligned(16))) _Float16 lsV[64 * 32];      // [d][key]
  __shared__ __attribute__((aligned(16))) _Float16 lsP[4][16 * 32];   // per-wave P tile
  const int tid = threadIdx.x;
  const int wave = tid >> 5, lane = tid & 31, g = lane >> 4, ln = lane & 15;
  const int bq0 = blockIdx.x * 64;
  const int h = blockIdx.y, b = blockIdx.z;
  const int q0w = bq0 + wave * 16;

  // Q fragments, K-dim = Dh = 64 -> two K=32 fragments, loaded once.
  v16h qa0, qa1;
  {
    const _Float16* qrow = qkv + (size_t)(b * SEQ + q0w + ln) * LDQKV + h * HDIM;
#pragma unroll
    for (int d = 0; d < 8; ++d) {
      int blk = d >> 2, i = d & 3;
      int k = blk * 16 + g * 8 + 2 * i;
      h2 p0 = *(const h2*)(qrow + k);
      qa0[2 * d] = p0.x; qa0[2 * d + 1] = p0.y;
      h2 p1 = *(const h2*)(qrow + 32 + k);
      qa1[2 * d] = p1.x; qa1[2 * d + 1] = p1.y;
    }
  }

  v8f acc[4] = {};
  float mrow[8], lrow[8];
#pragma unroll
  for (int v = 0; v < 8; ++v) { mrow[v] = -3.0e38f; lrow[v] = 0.0f; }

  const int nck = (bq0 + 64) >> 5;           // key chunks of 32
  for (int ck = 0; ck < nck; ++ck) {
    const int k0 = ck << 5;
#if USE_ASYNC_COPY
    // K tile: row-major copy -> async b128 (2 chunks/thread)
#pragma unroll
    for (int j = 0; j < 2; ++j) {
      int c = j * 128 + tid;                                   // 256 x 16B
      int key = c >> 3, d8 = (c & 7) * 8;
      async_copy16(qkv + (size_t)(b * SEQ + k0 + key) * LDQKV + EMBED + h * HDIM + d8,
                   &lsK[c * 8]);
    }
    // V tile: 2x b128 vector loads first, then 16 transposed LDS stores
    {
      int c0 = tid,       key0 = c0 >> 3, db0 = (c0 & 7) * 8;
      int c1 = 128 + tid, key1 = c1 >> 3, db1 = (c1 & 7) * 8;
      v8h v0 = *(const v8h*)&qkv[(size_t)(b * SEQ + k0 + key0) * LDQKV + 2 * EMBED + h * HDIM + db0];
      v8h v1 = *(const v8h*)&qkv[(size_t)(b * SEQ + k0 + key1) * LDQKV + 2 * EMBED + h * HDIM + db1];
#pragma unroll
      for (int i = 0; i < 8; ++i) lsV[(db0 + i) * 32 + key0] = v0[i];
#pragma unroll
      for (int i = 0; i < 8; ++i) lsV[(db1 + i) * 32 + key1] = v1[i];
    }
    __builtin_amdgcn_s_wait_asynccnt(0);
#else
    {
      int c0 = tid,       key0 = c0 >> 3, db0 = (c0 & 7) * 8;
      int c1 = 128 + tid, key1 = c1 >> 3, db1 = (c1 & 7) * 8;
      size_t r0 = (size_t)(b * SEQ + k0 + key0) * LDQKV + h * HDIM;
      size_t r1 = (size_t)(b * SEQ + k0 + key1) * LDQKV + h * HDIM;
      v8h k0v = *(const v8h*)&qkv[r0 + EMBED + db0];
      v8h k1v = *(const v8h*)&qkv[r1 + EMBED + db1];
      v8h v0  = *(const v8h*)&qkv[r0 + 2 * EMBED + db0];
      v8h v1  = *(const v8h*)&qkv[r1 + 2 * EMBED + db1];
      *(v8h*)&lsK[tid * 8] = k0v;
      *(v8h*)&lsK[(128 + tid) * 8] = k1v;
#pragma unroll
      for (int i = 0; i < 8; ++i) lsV[(db0 + i) * 32 + key0] = v0[i];
#pragma unroll
      for (int i = 0; i < 8; ++i) lsV[(db1 + i) * 32 + key1] = v1[i];
    }
#endif
    __syncthreads();

    if (k0 <= q0w + 15) {                    // wave-uniform: EXEC all-1s for WMMA
      // ---- scores: two 16x16 tiles over 32 keys
      v8f s0 = {}, s1 = {};
#pragma unroll
      for (int t = 0; t < 2; ++t) {
        v16h kb0, kb1;
        const _Float16* krow = &lsK[(t * 16 + ln) * 64];
#pragma unroll
        for (int d = 0; d < 8; ++d) {
          int k = (d >> 2) * 16 + g * 8 + 2 * (d & 3);
          h2 p0 = *(const h2*)(krow + k);
          kb0[2 * d] = p0.x; kb0[2 * d + 1] = p0.y;
          h2 p1 = *(const h2*)(krow + 32 + k);
          kb1[2 * d] = p1.x; kb1[2 * d + 1] = p1.y;
        }
        if (t == 0) { s0 = wmma_f16(qa0, kb0, s0); s0 = wmma_f16(qa1, kb1, s0); }
        else        { s1 = wmma_f16(qa0, kb0, s1); s1 = wmma_f16(qa1, kb1, s1); }
      }
      // ---- online softmax update (per row; uniform within each 16-lane half)
#pragma unroll
      for (int v = 0; v < 8; ++v) {
        const int q = q0w + v + 8 * g;
        if (q >= k0) {
          float a0 = s0[v] * 0.125f; if (k0 + ln      > q) a0 = -3.0e38f;
          float a1 = s1[v] * 0.125f; if (k0 + 16 + ln > q) a1 = -3.0e38f;
          float x = fmaxf(a0, a1);
#pragma unroll
          for (int off = 8; off; off >>= 1) x = fmaxf(x, __shfl_xor(x, off, 16));
          float mnew = fmaxf(mrow[v], x);
          float corr = __expf(mrow[v] - mnew);
          float p0 = __expf(a0 - mnew);
          float p1 = __expf(a1 - mnew);
          float rs = p0 + p1;
#pragma unroll
          for (int off = 8; off; off >>= 1) rs += __shfl_xor(rs, off, 16);
          lrow[v] = lrow[v] * corr + rs;
          mrow[v] = mnew;
#pragma unroll
          for (int nt = 0; nt < 4; ++nt) acc[nt][v] *= corr;
          lsP[wave][(v + 8 * g) * 32 + ln]      = (_Float16)p0;
          lsP[wave][(v + 8 * g) * 32 + 16 + ln] = (_Float16)p1;
        } else {
          lsP[wave][(v + 8 * g) * 32 + ln]      = (_Float16)0.f;
          lsP[wave][(v + 8 * g) * 32 + 16 + ln] = (_Float16)0.f;
        }
      }
      // ---- out += P(16x32) @ V(32x64) : 4 WMMAs over Dh tiles
      v16h pa;
#pragma unroll
      for (int d = 0; d < 8; ++d) {
        int blk = d >> 2, i = d & 3;
        int k = blk * 16 + g * 8 + 2 * i;
        h2 pp = *(const h2*)&lsP[wave][ln * 32 + k];
        pa[2 * d] = pp.x; pa[2 * d + 1] = pp.y;
      }
#pragma unroll
      for (int nt = 0; nt < 4; ++nt) {
        v16h vb;
#pragma unroll
        for (int d = 0; d < 8; ++d) {
          int k = (d >> 2) * 16 + g * 8 + 2 * (d & 3);
          h2 pv = *(const h2*)&lsV[(nt * 16 + ln) * 32 + k];
          vb[2 * d] = pv.x; vb[2 * d + 1] = pv.y;
        }
        acc[nt] = wmma_f16(pa, vb, acc[nt]);
      }
    }
    __syncthreads();
  }

#pragma unroll
  for (int v = 0; v < 8; ++v) {
    float inv = 1.0f / lrow[v];
    size_t base = (size_t)(b * SEQ + q0w + v + 8 * g) * EMBED + h * HDIM;
#pragma unroll
    for (int nt = 0; nt < 4; ++nt)
      ctx[base + nt * 16 + ln] = (_Float16)(acc[nt][v] * inv);
  }
}

// ---------------------------------------------------------------- launcher
extern "C" void kernel_launch(void* const* d_in, const int* in_sizes, int n_in,
                              void* d_out, int out_size, void* d_ws, size_t ws_size,
                              hipStream_t stream) {
  (void)in_sizes; (void)n_in; (void)out_size; (void)ws_size;
  const float* x     = (const float*)d_in[0];
  const float* w_qkv = (const float*)d_in[1];
  const float* b_qkv = (const float*)d_in[2];
  const float* w_out = (const float*)d_in[3];
  const float* b_out = (const float*)d_in[4];
  float* out = (float*)d_out;

  char* ws = (char*)d_ws;
  size_t off = 0;
  auto carve = [&](size_t elems) {
    _Float16* p = (_Float16*)(ws + off);
    off += ((elems * 2 + 255) & ~(size_t)255);
    return p;
  };
  _Float16* xh    = carve((size_t)BATCH * SEQ * EMBED);       // x in f16
  _Float16* wqkvh = carve((size_t)EMBED * 3 * EMBED);         // w_qkv f16
  _Float16* wouth = carve((size_t)EMBED * EMBED);             // w_out f16
  _Float16* qkvh  = carve((size_t)BATCH * SEQ * 3 * EMBED);   // QKV f16
  _Float16* ctxh  = carve((size_t)BATCH * SEQ * EMBED);       // attn output f16
  // total ~48 MB of workspace

  auto cast = [&](const float* src, _Float16* dst, size_t n) {
    unsigned blocks = (unsigned)((n / 8 + 255) / 256);
    cast_f32_f16<<<dim3(blocks), dim3(256), 0, stream>>>(src, dst, (int)n);
  };
  cast(x,     xh,    (size_t)BATCH * SEQ * EMBED);
  cast(w_qkv, wqkvh, (size_t)EMBED * 3 * EMBED);
  cast(w_out, wouth, (size_t)EMBED * EMBED);

  // QKV projection: [4096,1024] @ [1024,3072] + b_qkv -> f16
  wmma_gemm<<<dim3(3 * EMBED / 64, BATCH * SEQ / 128), dim3(256), 0, stream>>>(
      xh, wqkvh, b_qkv, qkvh, BATCH * SEQ, 3 * EMBED, EMBED, 1);

  // causal flash attention per (b, h, 64-query tile)
  attn_kernel<<<dim3(SEQ / 64, HEADS, BATCH), dim3(128), 0, stream>>>(qkvh, ctxh);

  // output projection: [4096,1024] @ [1024,1024] + b_out -> f32 d_out
  wmma_gemm<<<dim3(EMBED / 64, BATCH * SEQ / 128), dim3(256), 0, stream>>>(
      ctxh, wouth, b_out, out, BATCH * SEQ, EMBED, EMBED, 0);
}
